// DualAttnEncoderBlock_33440615366893
// MI455X (gfx1250) — compile-verified, matrix operand order checked by reading, not run on hardware
//
#include <hip/hip_runtime.h>
#include <hip/hip_bf16.h>

// ---------------------------------------------------------------------------
// Types for CDNA5 WMMA (gfx1250, wave32)
// ---------------------------------------------------------------------------
typedef __bf16 bf16_t;
typedef __attribute__((ext_vector_type(16))) __bf16 v16bf;
typedef __attribute__((ext_vector_type(8)))  __bf16 v8bf;
typedef __attribute__((ext_vector_type(8)))  float  v8f;

// Problem constants (B=2, T=1024, D=1024, HS=HR=R=8, HD=RD=64, DFF=4096)
#define BB   2
#define TT   1024
#define DD   1024
#define NH   8
#define HD   64
#define DFF  4096
#define MROWS (BB*TT)        // 2048
#define HDIM  (NH*HD)        // 512

// ---------------------------------------------------------------------------
// Fragment loader: 16-bit A/B operand layout per CDNA5 ISA 7.12.2.
// lane r = lane&15 supplies one row (A) / one column-as-row (B^T source);
// k elements: lanes 0-15 hold k in [0..7]∪[16..23], lanes 16-31 hold
// [8..15]∪[24..31]  =>  two contiguous 8-element (16B) loads per lane.
// ---------------------------------------------------------------------------
__device__ __forceinline__ v16bf load_frag(const bf16_t* base, int stride) {
  int lane = threadIdx.x & 31;
  int r = lane & 15, hi = lane >> 4;
  const bf16_t* p = base + (size_t)r * stride;
  v8bf lo = *(const v8bf*)(p + hi * 8);
  v8bf hh = *(const v8bf*)(p + 16 + hi * 8);
  v16bf out;
#pragma unroll
  for (int e = 0; e < 8; ++e) { out[e] = lo[e]; out[e + 8] = hh[e]; }
  return out;
}

__device__ __forceinline__ float half_reduce_max(float v) {
#pragma unroll
  for (int o = 1; o < 16; o <<= 1) v = fmaxf(v, __shfl_xor(v, o, 32));
  return v;
}
__device__ __forceinline__ float half_reduce_sum(float v) {
#pragma unroll
  for (int o = 1; o < 16; o <<= 1) v += __shfl_xor(v, o, 32);
  return v;
}

__device__ __forceinline__ v8f wmma_bf16(v16bf a, v16bf b, v8f c) {
  return __builtin_amdgcn_wmma_f32_16x16x32_bf16(false, a, false, b, (short)0, c,
                                                 false, false);
}

// ---------------------------------------------------------------------------
// Elementwise helpers
// ---------------------------------------------------------------------------
__global__ void cast_f32_bf16(const float* __restrict__ in, bf16_t* __restrict__ out, int n) {
  int i = blockIdx.x * blockDim.x + threadIdx.x;
  if (i < n) out[i] = (bf16_t)in[i];
}

// Weight cast + transpose: in[K][N] f32 -> out[N][K] bf16 (so GEMM B tiles
// stage into LDS with straight b128 stores, no scalar scatter).
__global__ void cast_transpose(const float* __restrict__ in, bf16_t* __restrict__ out,
                               int K, int N) {
  int i = blockIdx.x * blockDim.x + threadIdx.x;
  if (i >= K * N) return;
  int k = i / N, n = i - k * N;
  out[(size_t)n * K + k] = (bf16_t)in[i];
}

// One block (256 thr) per row of 1024; population variance like jnp.var.
__global__ void layernorm_to_bf16(const float* __restrict__ x, const float* __restrict__ g,
                                  const float* __restrict__ bta, bf16_t* __restrict__ out) {
  __shared__ float s1[256], s2[256];
  int row = blockIdx.x;
  const float* xr = x + (size_t)row * DD;
  float a = 0.f, b = 0.f;
  for (int i = threadIdx.x; i < DD; i += 256) { float v = xr[i]; a += v; b += v * v; }
  s1[threadIdx.x] = a; s2[threadIdx.x] = b;
  __syncthreads();
  for (int st = 128; st > 0; st >>= 1) {
    if (threadIdx.x < st) { s1[threadIdx.x] += s1[threadIdx.x + st]; s2[threadIdx.x] += s2[threadIdx.x + st]; }
    __syncthreads();
  }
  float mean = s1[0] * (1.f / DD);
  float var  = s2[0] * (1.f / DD) - mean * mean;
  float inv  = rsqrtf(var + 1e-5f);
  for (int i = threadIdx.x; i < DD; i += 256)
    out[(size_t)row * DD + i] = (bf16_t)(((xr[i] - mean) * inv) * g[i] + bta[i]);
}

// RoPE in place on a [B,T,8*64] bf16 buffer. One thread per (b,t,h,k) pair.
__global__ void rope_inplace(bf16_t* __restrict__ q, const float* __restrict__ cosb,
                             const float* __restrict__ sinb) {
  int idx = blockIdx.x * blockDim.x + threadIdx.x;
  if (idx >= BB * TT * NH * 32) return;
  int k = idx & 31;
  int h = (idx >> 5) & (NH - 1);
  int t = (idx >> 8) & (TT - 1);
  int b = idx >> 18;
  size_t base = ((size_t)(b * TT + t)) * HDIM + h * HD + 2 * k;
  float xr = (float)q[base], xi = (float)q[base + 1];
  float c = cosb[t * 32 + k], s = sinb[t * 32 + k];
  q[base]     = (bf16_t)(xr * c - xi * s);
  q[base + 1] = (bf16_t)(xr * s + xi * c);
}

// [b,t,h*64+d] -> [ (b*8+h)*64+d ][ t ]   (so P@V B-operand is contiguous)
__global__ void transpose_heads(const bf16_t* __restrict__ in, bf16_t* __restrict__ out) {
  int idx = blockIdx.x * blockDim.x + threadIdx.x;
  if (idx >= BB * TT * HDIM) return;
  int d512 = idx & (HDIM - 1);
  int t = (idx >> 9) & (TT - 1);
  int b = idx >> 19;
  int h = d512 >> 6, d = d512 & 63;
  out[(((size_t)(b * NH + h)) * HD + d) * TT + t] = in[idx];
}

// ra += einsum('bhir,hrd->bhid', att_rel, wr)
__global__ void add_relval(bf16_t* __restrict__ ra, const float* __restrict__ attrel,
                           const float* __restrict__ wr) {
  int idx = blockIdx.x * blockDim.x + threadIdx.x;
  if (idx >= BB * TT * HDIM) return;
  int d512 = idx & (HDIM - 1);
  int t = (idx >> 9) & (TT - 1);
  int b = idx >> 19;
  int h = d512 >> 6, d = d512 & 63;
  const float* ar = attrel + (((size_t)(b * TT + t)) * NH + h) * NH;
  float acc = (float)ra[idx];
#pragma unroll
  for (int r = 0; r < NH; ++r) acc += ar[r] * wr[(h * NH + r) * HD + d];
  ra[idx] = (bf16_t)acc;
}

// ---------------------------------------------------------------------------
// WMMA GEMM: C = act(A[MxK]bf16 @ B[KxN]bf16 + bias) (+ residual)
// B supplied pre-transposed (BT[N][K]). 256 threads = 8 waves.
// Block tile 128(M) x 128(N), K staged 64 deep, wave tile 32x64 =>
// 16 WMMAs per wave per stage against 24 ds_load_b128 (A frag reused 4x,
// B frag reused 2x). Next stage's global tiles prefetched into registers;
// last stage peeled so the steady-state loop is branch-free.
// ---------------------------------------------------------------------------
#define KS 64
__global__ __launch_bounds__(256) void wmma_gemm(
    const bf16_t* __restrict__ A, int lda, const bf16_t* __restrict__ BT,
    const float* __restrict__ bias, const float* __restrict__ res, int ldr,
    float* __restrict__ Cf, int ldcf, bf16_t* __restrict__ Cb, int ldcb,
    int K, int relu) {
  __shared__ bf16_t As[128][KS];
  __shared__ bf16_t Bs[128][KS];
  int tid = threadIdx.x;
  int lane = tid & 31, wave = tid >> 5;
  int hi = lane >> 4, ln = lane & 15;
  int m0 = blockIdx.y * 128, n0 = blockIdx.x * 128;
  int wm = (wave >> 1) * 32;   // 0,32,64,96
  int wn = (wave & 1) * 64;    // 0,64
  v8f acc[2][4];
#pragma unroll
  for (int mi = 0; mi < 2; ++mi)
#pragma unroll
    for (int c = 0; c < 4; ++c)
#pragma unroll
      for (int e = 0; e < 8; ++e) acc[mi][c][e] = 0.f;

  int r8 = tid >> 3, kb = (tid & 7) * 8;   // staging slot: rows r8+32i, k-block kb
  const bf16_t* Ag = A  + (size_t)(m0 + r8) * lda + kb;
  const bf16_t* Bg = BT + (size_t)(n0 + r8) * K + kb;

  v8bf ar[4], br[4];
  auto gload = [&](int k0) {
#pragma unroll
    for (int i = 0; i < 4; ++i) {
      ar[i] = *(const v8bf*)(Ag + (size_t)(32 * i) * lda + k0);
      br[i] = *(const v8bf*)(Bg + (size_t)(32 * i) * K + k0);
    }
  };
  auto stage_store = [&]() {
#pragma unroll
    for (int i = 0; i < 4; ++i) {
      *(v8bf*)&As[r8 + 32 * i][kb] = ar[i];
      *(v8bf*)&Bs[r8 + 32 * i][kb] = br[i];
    }
  };
  auto compute = [&]() {
#pragma unroll
    for (int kk = 0; kk < KS; kk += 32) {
      v16bf a0 = load_frag(&As[wm][kk], KS);
      v16bf a1 = load_frag(&As[wm + 16][kk], KS);
      v16bf b0 = load_frag(&Bs[wn][kk], KS);
      v16bf b1 = load_frag(&Bs[wn + 16][kk], KS);
      v16bf b2 = load_frag(&Bs[wn + 32][kk], KS);
      v16bf b3 = load_frag(&Bs[wn + 48][kk], KS);
      acc[0][0] = wmma_bf16(a0, b0, acc[0][0]);
      acc[0][1] = wmma_bf16(a0, b1, acc[0][1]);
      acc[0][2] = wmma_bf16(a0, b2, acc[0][2]);
      acc[0][3] = wmma_bf16(a0, b3, acc[0][3]);
      acc[1][0] = wmma_bf16(a1, b0, acc[1][0]);
      acc[1][1] = wmma_bf16(a1, b1, acc[1][1]);
      acc[1][2] = wmma_bf16(a1, b2, acc[1][2]);
      acc[1][3] = wmma_bf16(a1, b3, acc[1][3]);
    }
  };

  gload(0);
  int k0 = 0;
  for (; k0 + KS < K; k0 += KS) {
    stage_store();
    __syncthreads();
    gload(k0 + KS);          // prefetch next stage during compute
    compute();
    __syncthreads();
  }
  stage_store();             // peeled final stage (no prefetch, no branch)
  __syncthreads();
  compute();

#pragma unroll
  for (int mi = 0; mi < 2; ++mi)
#pragma unroll
    for (int c = 0; c < 4; ++c) {
      int n = n0 + wn + c * 16 + ln;
      float bval = bias ? bias[n] : 0.f;
#pragma unroll
      for (int v = 0; v < 8; ++v) {
        int m = m0 + wm + mi * 16 + 8 * hi + v;
        float val = acc[mi][c][v] + bval;
        if (relu) val = fmaxf(val, 0.f);
        if (res) val += res[(size_t)m * ldr + n];
        if (Cf) Cf[(size_t)m * ldcf + n] = val;
        if (Cb) Cb[(size_t)m * ldcb + n] = (bf16_t)val;
      }
    }
}

// ---------------------------------------------------------------------------
// rel[b,r,i,j] = (rq[b,r,i,:] . rk[b,r,j,:]) / sqrt(RD), bf16 out.
// One wave per (i-tile, r, b); stream j in steps of 32.
// ---------------------------------------------------------------------------
__global__ __launch_bounds__(32) void rel_scores(const bf16_t* __restrict__ RQ,
                                                 const bf16_t* __restrict__ RK,
                                                 bf16_t* __restrict__ rel, float scale) {
  int lane = threadIdx.x, hi = lane >> 4, ln = lane & 15;
  int i0 = blockIdx.x * 16, r = blockIdx.y, b = blockIdx.z;
  const bf16_t* qb = RQ + ((size_t)(b * TT + i0)) * HDIM + r * HD;
  v16bf a0 = load_frag(qb, HDIM);
  v16bf a1 = load_frag(qb + 32, HDIM);
  bf16_t* ob = rel + (((size_t)(b * NH + r)) * TT + i0 + 8 * hi) * TT;
  for (int j0 = 0; j0 < TT; j0 += 32) {
    const bf16_t* kb = RK + ((size_t)(b * TT + j0)) * HDIM + r * HD;
    v16bf k00 = load_frag(kb, HDIM);
    v16bf k01 = load_frag(kb + 32, HDIM);
    v16bf k10 = load_frag(kb + 16 * HDIM, HDIM);
    v16bf k11 = load_frag(kb + 16 * HDIM + 32, HDIM);
    v8f S0, S1;
#pragma unroll
    for (int e = 0; e < 8; ++e) { S0[e] = 0.f; S1[e] = 0.f; }
    S0 = wmma_bf16(a0, k00, S0);
    S0 = wmma_bf16(a1, k01, S0);
    S1 = wmma_bf16(a0, k10, S1);
    S1 = wmma_bf16(a1, k11, S1);
#pragma unroll
    for (int v = 0; v < 8; ++v) {
      ob[(size_t)v * TT + j0 + ln]      = (bf16_t)(S0[v] * scale);
      ob[(size_t)v * TT + j0 + 16 + ln] = (bf16_t)(S1[v] * scale);
    }
  }
}

// ---------------------------------------------------------------------------
// Flash attention, one wave per (16-row Q tile, head, batch).
// REL=1 additionally accumulates att_rel[b,i,h,r] = sum_j aa*rel (normalized),
// reading precomputed bf16 rel tiles from global.
// ---------------------------------------------------------------------------
template <int REL>
__global__ __launch_bounds__(32) void flash_attn(
    const bf16_t* __restrict__ Q, const bf16_t* __restrict__ Kb,
    const bf16_t* __restrict__ VT, bf16_t* __restrict__ O,
    const bf16_t* __restrict__ rel, float* __restrict__ attrel, float scale) {
  __shared__ bf16_t Pt[16][32];
  int lane = threadIdx.x, hi = lane >> 4, ln = lane & 15;
  int i0 = blockIdx.x * 16, h = blockIdx.y, b = blockIdx.z;

  const bf16_t* qb = Q + ((size_t)(b * TT + i0)) * HDIM + h * HD;
  v16bf a0 = load_frag(qb, HDIM);
  v16bf a1 = load_frag(qb + 32, HDIM);

  float m_s[8], l_s[8], accR[8];
  v8f accO[4];
#pragma unroll
  for (int v = 0; v < 8; ++v) { m_s[v] = -1e30f; l_s[v] = 0.f; accR[v] = 0.f; }
#pragma unroll
  for (int c = 0; c < 4; ++c)
#pragma unroll
    for (int e = 0; e < 8; ++e) accO[c][e] = 0.f;

  for (int j0 = 0; j0 < TT; j0 += 32) {
    const bf16_t* kb = Kb + ((size_t)(b * TT + j0)) * HDIM + h * HD;
    v16bf k00 = load_frag(kb, HDIM);
    v16bf k01 = load_frag(kb + 32, HDIM);
    v16bf k10 = load_frag(kb + 16 * HDIM, HDIM);
    v16bf k11 = load_frag(kb + 16 * HDIM + 32, HDIM);
    v8f S0, S1;
#pragma unroll
    for (int e = 0; e < 8; ++e) { S0[e] = 0.f; S1[e] = 0.f; }
    S0 = wmma_bf16(a0, k00, S0);
    S0 = wmma_bf16(a1, k01, S0);
    S1 = wmma_bf16(a0, k10, S1);
    S1 = wmma_bf16(a1, k11, S1);

    float p0[8], p1[8], corr[8];
#pragma unroll
    for (int v = 0; v < 8; ++v) {
      float s0 = S0[v] * scale, s1 = S1[v] * scale;
      float mx = half_reduce_max(fmaxf(s0, s1));
      float mnew = fmaxf(m_s[v], mx);
      float c = __expf(m_s[v] - mnew);
      p0[v] = __expf(s0 - mnew);
      p1[v] = __expf(s1 - mnew);
      float rs = half_reduce_sum(p0[v] + p1[v]);
      l_s[v] = l_s[v] * c + rs;
      m_s[v] = mnew;
      corr[v] = c;
#pragma unroll
      for (int cc = 0; cc < 4; ++cc) accO[cc][v] *= c;
      Pt[8 * hi + v][ln]      = (bf16_t)p0[v];
      Pt[8 * hi + v][16 + ln] = (bf16_t)p1[v];
    }

    if (REL) {
#pragma unroll
      for (int v = 0; v < 8; ++v) accR[v] *= corr[v];
#pragma unroll
      for (int r = 0; r < NH; ++r) {
        const bf16_t* rb =
            rel + (((size_t)(b * NH + r)) * TT + (i0 + 8 * hi)) * TT + j0 + ln;
#pragma unroll
        for (int v = 0; v < 8; ++v) {
          float r0 = (float)rb[(size_t)v * TT];
          float r1 = (float)rb[(size_t)v * TT + 16];
          float t = half_reduce_sum(p0[v] * r0 + p1[v] * r1);
          if ((lane & 7) == r) accR[v] += t;
        }
      }
    }

    __syncthreads();   // P-tile visibility (S_NOP-cost barrier for 1-wave WG)
    v16bf pa = load_frag(&Pt[0][0], 32);
    const bf16_t* vb = VT + (((size_t)(b * NH + h)) * HD) * TT + j0;
    v16bf v0 = load_frag(vb, TT);
    v16bf v1 = load_frag(vb + (size_t)16 * TT, TT);
    v16bf v2 = load_frag(vb + (size_t)32 * TT, TT);
    v16bf v3 = load_frag(vb + (size_t)48 * TT, TT);
    accO[0] = wmma_bf16(pa, v0, accO[0]);
    accO[1] = wmma_bf16(pa, v1, accO[1]);
    accO[2] = wmma_bf16(pa, v2, accO[2]);
    accO[3] = wmma_bf16(pa, v3, accO[3]);
    __syncthreads();
  }

  float inv[8];
#pragma unroll
  for (int v = 0; v < 8; ++v) inv[v] = 1.f / l_s[v];
#pragma unroll
  for (int c = 0; c < 4; ++c)
#pragma unroll
    for (int v = 0; v < 8; ++v) {
      int m = i0 + 8 * hi + v;
      O[((size_t)(b * TT + m)) * HDIM + h * HD + c * 16 + ln] =
          (bf16_t)(accO[c][v] * inv[v]);
    }
  if (REL && ln < 8) {
#pragma unroll
    for (int v = 0; v < 8; ++v) {
      int m = i0 + 8 * hi + v;
      attrel[(((size_t)(b * TT + m)) * NH + h) * NH + (lane & 7)] = accR[v] * inv[v];
    }
  }
}

// ---------------------------------------------------------------------------
// Host launcher
// ---------------------------------------------------------------------------
static void gemm(const bf16_t* A, int lda, const bf16_t* BT, const float* bias,
                 const float* res, int ldr, float* Cf, int ldcf, bf16_t* Cb, int ldcb,
                 int M, int N, int K, int relu, hipStream_t s) {
  dim3 g(N / 128, M / 128);
  wmma_gemm<<<g, 256, 0, s>>>(A, lda, BT, bias, res, ldr, Cf, ldcf, Cb, ldcb, K, relu);
}

extern "C" void kernel_launch(void* const* d_in, const int* in_sizes, int n_in,
                              void* d_out, int out_size, void* d_ws, size_t ws_size,
                              hipStream_t stream) {
  const float* x        = (const float*)d_in[0];
  const float* symbols  = (const float*)d_in[1];
  const float* fcos     = (const float*)d_in[2];
  const float* fsin     = (const float*)d_in[3];
  const float* g1       = (const float*)d_in[4];
  const float* beta1    = (const float*)d_in[5];
  const float* g2       = (const float*)d_in[6];
  const float* beta2    = (const float*)d_in[7];
  const float* wq_sa    = (const float*)d_in[8];
  const float* bq_sa    = (const float*)d_in[9];
  const float* wk_sa    = (const float*)d_in[10];
  const float* bk_sa    = (const float*)d_in[11];
  const float* wv_sa    = (const float*)d_in[12];
  const float* bv_sa    = (const float*)d_in[13];
  const float* wo_sa    = (const float*)d_in[14];
  const float* bo_sa    = (const float*)d_in[15];
  const float* wq_ra    = (const float*)d_in[16];
  const float* bq_ra    = (const float*)d_in[17];
  const float* wk_ra    = (const float*)d_in[18];
  const float* bk_ra    = (const float*)d_in[19];
  const float* wv_sym   = (const float*)d_in[20];
  const float* bv_sym   = (const float*)d_in[21];
  const float* wqr      = (const float*)d_in[22];
  const float* wkr      = (const float*)d_in[23];
  const float* wr       = (const float*)d_in[24];
  const float* wo_ra    = (const float*)d_in[25];
  const float* bo_ra    = (const float*)d_in[26];
  const float* w1       = (const float*)d_in[27];
  const float* b1f      = (const float*)d_in[28];
  const float* w2       = (const float*)d_in[29];
  const float* b2f      = (const float*)d_in[30];
  float* out = (float*)d_out;

  char* w = (char*)d_ws;
  size_t o = 0;
  auto take = [&](size_t bytes) -> char* {
    char* p = w + o;
    o += (bytes + 255) & ~(size_t)255;
    return p;
  };
  // bf16 transposed weights (BT[N][K])
  bf16_t* wq_sa_b = (bf16_t*)take((size_t)DD * HDIM * 2);
  bf16_t* wk_sa_b = (bf16_t*)take((size_t)DD * HDIM * 2);
  bf16_t* wv_sa_b = (bf16_t*)take((size_t)DD * HDIM * 2);
  bf16_t* wq_ra_b = (bf16_t*)take((size_t)DD * HDIM * 2);
  bf16_t* wk_ra_b = (bf16_t*)take((size_t)DD * HDIM * 2);
  bf16_t* wv_sy_b = (bf16_t*)take((size_t)DD * HDIM * 2);
  bf16_t* wqr_b   = (bf16_t*)take((size_t)DD * HDIM * 2);
  bf16_t* wkr_b   = (bf16_t*)take((size_t)DD * HDIM * 2);
  bf16_t* wo_sa_b = (bf16_t*)take((size_t)HDIM * HDIM * 2);
  bf16_t* wo_ra_b = (bf16_t*)take((size_t)HDIM * HDIM * 2);
  bf16_t* w1_b    = (bf16_t*)take((size_t)DD * DFF * 2);
  bf16_t* w2_b    = (bf16_t*)take((size_t)DFF * DD * 2);
  // activations
  bf16_t* h_b   = (bf16_t*)take((size_t)MROWS * DD * 2);
  bf16_t* sym_b = (bf16_t*)take((size_t)MROWS * DD * 2);
  bf16_t* q_sa  = (bf16_t*)take((size_t)MROWS * HDIM * 2);
  bf16_t* k_sa  = (bf16_t*)take((size_t)MROWS * HDIM * 2);
  bf16_t* v_sa  = (bf16_t*)take((size_t)MROWS * HDIM * 2);
  bf16_t* qa    = (bf16_t*)take((size_t)MROWS * HDIM * 2);
  bf16_t* ka    = (bf16_t*)take((size_t)MROWS * HDIM * 2);
  bf16_t* sv    = (bf16_t*)take((size_t)MROWS * HDIM * 2);
  bf16_t* rq    = (bf16_t*)take((size_t)MROWS * HDIM * 2);
  bf16_t* rk    = (bf16_t*)take((size_t)MROWS * HDIM * 2);
  bf16_t* vT    = (bf16_t*)take((size_t)MROWS * HDIM * 2);
  bf16_t* svT   = (bf16_t*)take((size_t)MROWS * HDIM * 2);
  bf16_t* saout = (bf16_t*)take((size_t)MROWS * HDIM * 2);
  bf16_t* rain  = (bf16_t*)take((size_t)MROWS * HDIM * 2);
  float*  attrl = (float*)take((size_t)MROWS * NH * NH * 4);
  bf16_t* relb  = (bf16_t*)take((size_t)BB * NH * TT * TT * 2);
  float*  xnew  = (float*)take((size_t)MROWS * DD * 4);
  bf16_t* h2_b  = (bf16_t*)take((size_t)MROWS * DD * 2);
  bf16_t* ff1_b = (bf16_t*)take((size_t)MROWS * DFF * 2);
  (void)ws_size; (void)n_in; (void)in_sizes; (void)out_size;

  auto castT = [&](const float* src, bf16_t* dst, int K, int N) {
    cast_transpose<<<(K * N + 255) / 256, 256, 0, stream>>>(src, dst, K, N);
  };
  castT(wq_sa, wq_sa_b, DD, HDIM);
  castT(wk_sa, wk_sa_b, DD, HDIM);
  castT(wv_sa, wv_sa_b, DD, HDIM);
  castT(wq_ra, wq_ra_b, DD, HDIM);
  castT(wk_ra, wk_ra_b, DD, HDIM);
  castT(wv_sym, wv_sy_b, DD, HDIM);
  castT(wqr, wqr_b, DD, HDIM);
  castT(wkr, wkr_b, DD, HDIM);
  castT(wo_sa, wo_sa_b, HDIM, HDIM);
  castT(wo_ra, wo_ra_b, HDIM, HDIM);
  castT(w1, w1_b, DD, DFF);
  castT(w2, w2_b, DFF, DD);
  cast_f32_bf16<<<(MROWS * DD + 255) / 256, 256, 0, stream>>>(symbols, sym_b, MROWS * DD);

  // LN1
  layernorm_to_bf16<<<MROWS, 256, 0, stream>>>(x, g1, beta1, h_b);

  // Projections (M=2048, K=1024, N=512)
  gemm(h_b, DD, wq_sa_b, bq_sa, nullptr, 0, nullptr, 0, q_sa, HDIM, MROWS, HDIM, DD, 0, stream);
  gemm(h_b, DD, wk_sa_b, bk_sa, nullptr, 0, nullptr, 0, k_sa, HDIM, MROWS, HDIM, DD, 0, stream);
  gemm(h_b, DD, wv_sa_b, bv_sa, nullptr, 0, nullptr, 0, v_sa, HDIM, MROWS, HDIM, DD, 0, stream);
  gemm(h_b, DD, wq_ra_b, bq_ra, nullptr, 0, nullptr, 0, qa,   HDIM, MROWS, HDIM, DD, 0, stream);
  gemm(h_b, DD, wk_ra_b, bk_ra, nullptr, 0, nullptr, 0, ka,   HDIM, MROWS, HDIM, DD, 0, stream);
  gemm(sym_b, DD, wv_sy_b, bv_sym, nullptr, 0, nullptr, 0, sv, HDIM, MROWS, HDIM, DD, 0, stream);
  gemm(h_b, DD, wqr_b, nullptr, nullptr, 0, nullptr, 0, rq, HDIM, MROWS, HDIM, DD, 0, stream);
  gemm(h_b, DD, wkr_b, nullptr, nullptr, 0, nullptr, 0, rk, HDIM, MROWS, HDIM, DD, 0, stream);

  // RoPE on q/k of both attentions
  int nrp = BB * TT * NH * 32;
  rope_inplace<<<(nrp + 255) / 256, 256, 0, stream>>>(q_sa, fcos, fsin);
  rope_inplace<<<(nrp + 255) / 256, 256, 0, stream>>>(k_sa, fcos, fsin);
  rope_inplace<<<(nrp + 255) / 256, 256, 0, stream>>>(qa, fcos, fsin);
  rope_inplace<<<(nrp + 255) / 256, 256, 0, stream>>>(ka, fcos, fsin);

  // Transpose V / SV for the P@V B-operand
  int ntr = MROWS * HDIM;
  transpose_heads<<<(ntr + 255) / 256, 256, 0, stream>>>(v_sa, vT);
  transpose_heads<<<(ntr + 255) / 256, 256, 0, stream>>>(sv, svT);

  // rel = rq rk^T / sqrt(RD)
  rel_scores<<<dim3(TT / 16, NH, BB), 32, 0, stream>>>(rq, rk, relb, 0.125f);

  // Flash attention (self) and (relational, with att_rel accumulation)
  const float scale = 0.125f;  // 1/sqrt(64)
  flash_attn<0><<<dim3(TT / 16, NH, BB), 32, 0, stream>>>(q_sa, k_sa, vT, saout,
                                                          nullptr, nullptr, scale);
  flash_attn<1><<<dim3(TT / 16, NH, BB), 32, 0, stream>>>(qa, ka, svT, rain, relb,
                                                          attrl, scale);
  // rain += att_rel @ wr
  add_relval<<<(ntr + 255) / 256, 256, 0, stream>>>(rain, attrl, wr);

  // Output projections + residual into xnew (concat: sa -> cols 0..511, ra -> 512..1023)
  gemm(saout, HDIM, wo_sa_b, bo_sa, x, DD, xnew, DD, nullptr, 0,
       MROWS, HDIM, HDIM, 0, stream);
  gemm(rain, HDIM, wo_ra_b, bo_ra, x + HDIM, DD, xnew + HDIM, DD, nullptr, 0,
       MROWS, HDIM, HDIM, 0, stream);

  // LN2 + FFN + residual
  layernorm_to_bf16<<<MROWS, 256, 0, stream>>>(xnew, g2, beta2, h2_b);
  gemm(h2_b, DD, w1_b, b1f, nullptr, 0, nullptr, 0, ff1_b, DFF,
       MROWS, DFF, DD, 1, stream);
  gemm(ff1_b, DFF, w2_b, b2f, xnew, DD, out, DD, nullptr, 0,
       MROWS, DD, DFF, 0, stream);
}
